// LeoSLM_84559316123683
// MI455X (gfx1250) — matrix operand order repaired and checked by reading, not compiled, not run on hardware
//
#include <hip/hip_runtime.h>
#include <cstdint>
#include <cstddef>

typedef __bf16 bf16;
typedef __attribute__((ext_vector_type(16))) __bf16 v16bf;
typedef __attribute__((ext_vector_type(8)))  float  v8f;
typedef int v4i_ __attribute__((__vector_size__(16)));

#define T_   2048
#define D_   2560
#define NH_  20
#define DH_  128
#define DR_  64
#define DQK_ 192
#define CQ_  768
#define CKV_ 512
#define NE_  8
#define TOPK_ 2
#define FE_  1024
#define NSLOT_ (T_*TOPK_)

// ---------------------------------------------------------------------------
// Async global->LDS copy (gfx1250 GLOBAL_LOAD_ASYNC_TO_LDS_B128), with a
// synchronous fallback if the builtin is unavailable on this toolchain.
// Builtin signature (from probe): (int4 AS1*, int4 AS3*, imm offset, imm cpol)
// ---------------------------------------------------------------------------
#if defined(__AMDGCN__) && __has_builtin(__builtin_amdgcn_global_load_async_to_lds_b128)
#define USE_ASYNC_LDS 1
#else
#define USE_ASYNC_LDS 0
#endif

__device__ __forceinline__ void async_b128(const void* g, void* l) {
#if USE_ASYNC_LDS
  __builtin_amdgcn_global_load_async_to_lds_b128(
      (__attribute__((address_space(1))) v4i_*)g,
      (__attribute__((address_space(3))) v4i_*)l, 0, 0);
#else
  *reinterpret_cast<uint4*>(l) = *reinterpret_cast<const uint4*>(g);
#endif
}

__device__ __forceinline__ void wait_async() {
#if USE_ASYNC_LDS
  asm volatile("s_wait_asynccnt 0" ::: "memory");
#endif
}

// ---------------------------------------------------------------------------
// WMMA helpers (bf16 16x16x32, f32 accum).
// ---------------------------------------------------------------------------
__device__ __forceinline__ v8f wmma_bf16(v16bf a, v16bf b, v8f c) {
  return __builtin_amdgcn_wmma_f32_16x16x32_bf16(false, a, false, b, (short)0, c,
                                                 false, false);
}

__device__ __forceinline__ bf16 u16_to_bf(unsigned short u) {
  return __builtin_bit_cast(bf16, u);
}

// A fragment: lane L holds row M = L&15; VGPR pairs hold K per ISA table.
__device__ __forceinline__ v16bf load_frag_a(const bf16* p, int stride) {
  const int lane = threadIdx.x & 31;
  const int row  = lane & 15;
  const int hi   = lane >> 4;
  const bf16* rp = p + row * stride + hi * 8;
  v16bf v;
#pragma unroll
  for (int i = 0; i < 4; ++i) {
    uint32_t u = *reinterpret_cast<const uint32_t*>(rp + 2 * i);
    v[2 * i]     = u16_to_bf((unsigned short)(u & 0xffffu));
    v[2 * i + 1] = u16_to_bf((unsigned short)(u >> 16));
  }
#pragma unroll
  for (int i = 0; i < 4; ++i) {
    uint32_t u = *reinterpret_cast<const uint32_t*>(rp + 16 + 2 * i);
    v[8 + 2 * i] = u16_to_bf((unsigned short)(u & 0xffffu));
    v[9 + 2 * i] = u16_to_bf((unsigned short)(u >> 16));
  }
  return v;
}

// B fragment from LDS stored as [N][K]: lane holds column N = L&15,
// K order per ISA table (same as A fragment).
__device__ __forceinline__ v16bf load_frag_b(const bf16* p, int stride) {
  const int lane = threadIdx.x & 31;
  const int n    = lane & 15;
  const int hi   = lane >> 4;
  const bf16* rp = p + n * stride + hi * 8;
  v16bf v;
#pragma unroll
  for (int i = 0; i < 8; ++i) {
    uint32_t u = *reinterpret_cast<const uint32_t*>(rp + (i >> 2) * 16 + 2 * (i & 3));
    v[2 * i]     = u16_to_bf((unsigned short)(u & 0xffffu));
    v[2 * i + 1] = u16_to_bf((unsigned short)(u >> 16));
  }
  return v;
}

// ---------------------------------------------------------------------------
// fp32 -> bf16 bulk conversion (vectorized x4)
// ---------------------------------------------------------------------------
__global__ __launch_bounds__(256) void k_f2bf(const float* __restrict__ in,
                                              bf16* __restrict__ out, int n4) {
  const int i = blockIdx.x * 256 + threadIdx.x;
  if (i < n4) {
    const float4 f = reinterpret_cast<const float4*>(in)[i];
    uint2 u;
    u.x = (uint32_t)__builtin_bit_cast(unsigned short, (bf16)f.x) |
          ((uint32_t)__builtin_bit_cast(unsigned short, (bf16)f.y) << 16);
    u.y = (uint32_t)__builtin_bit_cast(unsigned short, (bf16)f.z) |
          ((uint32_t)__builtin_bit_cast(unsigned short, (bf16)f.w) << 16);
    reinterpret_cast<uint2*>(out)[i] = u;
  }
}

// ---------------------------------------------------------------------------
// RMSNorm: optional fp32 and/or bf16 outputs (either may be null)
// ---------------------------------------------------------------------------
__global__ __launch_bounds__(256) void k_rms(const float* __restrict__ in,
                                             const float* __restrict__ g,
                                             float* __restrict__ outf,
                                             bf16* __restrict__ outb, int ncols) {
  const int row = blockIdx.x;
  const float* r = in + (size_t)row * ncols;
  __shared__ float red[256];
  float ss = 0.f;
  for (int c = threadIdx.x; c < ncols; c += 256) { float v = r[c]; ss += v * v; }
  red[threadIdx.x] = ss;
  __syncthreads();
  for (int s = 128; s > 0; s >>= 1) {
    if (threadIdx.x < s) red[threadIdx.x] += red[threadIdx.x + s];
    __syncthreads();
  }
  const float inv = rsqrtf(red[0] / (float)ncols + 1e-6f);
  for (int c = threadIdx.x; c < ncols; c += 256) {
    const float v = r[c] * g[c] * inv;
    if (outf) outf[(size_t)row * ncols + c] = v;
    if (outb) outb[(size_t)row * ncols + c] = (bf16)v;
  }
}

// ---------------------------------------------------------------------------
// Unified double-buffered bf16 WMMA GEMM, tile 128x128x64, 8 waves (2x4).
// A-tiles stream in via async-to-LDS; B-tiles are transposed into [N][K] LDS.
//  mode 0: C[MxN] = A[MxK]*B[KxN] (+addm)
//  mode 1: MoE up/gate: A rows gathered via tok[], C stored by slot
//  mode 2: MoE down:    A rows by slot, atomicAdd(w*val) scattered by token
// ---------------------------------------------------------------------------
#define LDT2 72
__global__ __launch_bounds__(256) void k_gemm_bf(
    const bf16* __restrict__ A, const bf16* __restrict__ Ball,
    const float* __restrict__ addm, float* __restrict__ C,
    int M, int N, int K,
    const int* __restrict__ tok, const float* __restrict__ wts,
    const int* __restrict__ cnts, const int* __restrict__ offs, int mode) {
  __shared__ bf16 As[2][128 * LDT2];
  __shared__ bf16 Bs[2][128 * LDT2];
  const int tid = threadIdx.x;
  const int m0 = blockIdx.y * 128, n0 = blockIdx.x * 128;
  const bf16* B = Ball;
  int cnt = M, seg = 0;
  if (mode != 0) {
    const int e = blockIdx.z;
    cnt = cnts[e];
    seg = offs[e];
    if (m0 >= cnt) return;
    B += (size_t)e * K * N;
  }

  // Per-thread A-copy descriptors: 4 x 16B segments of the 128x64 bf16 tile.
  size_t a_off[4];
  int a_lds[4];
  bool a_ok[4];
#pragma unroll
  for (int p = 0; p < 4; ++p) {
    const int s = tid + p * 256;
    const int row = s >> 3, sir = s & 7;
    const int m = m0 + row;
    int grow = m;
    bool ok;
    if (mode == 0) {
      ok = (m < M);
    } else if (mode == 1) {
      ok = (m < cnt);
      grow = ok ? tok[seg + m] : 0;
    } else {
      ok = (m < cnt);
      grow = seg + m;
    }
    a_ok[p] = ok;
    a_off[p] = (size_t)grow * K + sir * 8;
    a_lds[p] = row * LDT2 + sir * 8;
  }
  const int bc8 = (tid & 15) * 8;
  const int brw = tid >> 4;

  auto load_tile = [&](int kb, int buf) {
#pragma unroll
    for (int p = 0; p < 4; ++p)
      if (a_ok[p]) async_b128(A + a_off[p] + kb, &As[buf][a_lds[p]]);
#pragma unroll
    for (int pp = 0; pp < 4; ++pp) {
      const int r = brw + pp * 16;
      uint4 v = {0u, 0u, 0u, 0u};
      if (n0 + bc8 < N)
        v = *reinterpret_cast<const uint4*>(B + (size_t)(kb + r) * N + n0 + bc8);
      const bf16* pv = reinterpret_cast<const bf16*>(&v);
#pragma unroll
      for (int j = 0; j < 8; ++j) Bs[buf][(bc8 + j) * LDT2 + r] = pv[j];
    }
  };

  const int wid = tid >> 5, lane = tid & 31;
  const int wm = wid >> 2, wn = wid & 3;
  v8f acc[4][2];
#pragma unroll
  for (int i = 0; i < 4; ++i)
#pragma unroll
    for (int j = 0; j < 2; ++j) acc[i][j] = v8f{0, 0, 0, 0, 0, 0, 0, 0};

  const int nk = K >> 6;
  load_tile(0, 0);
  wait_async();
  __syncthreads();
  for (int it = 0; it < nk; ++it) {
    const int cur = it & 1;
    if (it + 1 < nk) load_tile((it + 1) << 6, cur ^ 1);
#pragma unroll
    for (int ks = 0; ks < 2; ++ks) {
      v16bf af[4], bfv[2];
#pragma unroll
      for (int i = 0; i < 4; ++i)
        af[i] = load_frag_a(&As[cur][(wm * 64 + i * 16) * LDT2 + ks * 32], LDT2);
#pragma unroll
      for (int j = 0; j < 2; ++j)
        bfv[j] = load_frag_b(&Bs[cur][(wn * 32 + j * 16) * LDT2 + ks * 32], LDT2);
#pragma unroll
      for (int i = 0; i < 4; ++i)
#pragma unroll
        for (int j = 0; j < 2; ++j) acc[i][j] = wmma_bf16(af[i], bfv[j], acc[i][j]);
    }
    wait_async();
    __syncthreads();
  }

  const int cn = lane & 15, hi = lane >> 4;
#pragma unroll
  for (int i = 0; i < 4; ++i)
#pragma unroll
    for (int j = 0; j < 2; ++j) {
      const int gn = n0 + wn * 32 + j * 16 + cn;
      if (gn >= N) continue;
#pragma unroll
      for (int r = 0; r < 8; ++r) {
        const int m = m0 + wm * 64 + i * 16 + r + hi * 8;
        float v = acc[i][j][r];
        if (mode == 0) {
          if (m < M) {
            if (addm) v += addm[(size_t)m * N + gn];
            C[(size_t)m * N + gn] = v;
          }
        } else if (m < cnt) {
          if (mode == 1) {
            C[(size_t)(seg + m) * N + gn] = v;
          } else {
            atomicAdd(&C[(size_t)tok[seg + m] * N + gn], wts[seg + m] * v);
          }
        }
      }
    }
}

// ---------------------------------------------------------------------------
// Pack Q/K/V to bf16 per head; apply RoPE and fold SCALE into Q.
// ---------------------------------------------------------------------------
__global__ __launch_bounds__(256) void k_pack(
    const float* __restrict__ q, const float* __restrict__ kv,
    const float* __restrict__ kr, const float* __restrict__ cs,
    const float* __restrict__ sn, bf16* __restrict__ Qp, bf16* __restrict__ Kp,
    bf16* __restrict__ Vp) {
  const int t = blockIdx.x;
  const int tid = threadIdx.x;
  const float SC = rsqrtf((float)DQK_);
  if (tid >= DQK_) return;
  for (int h = 0; h < NH_; ++h) {
    const size_t qk = ((size_t)h * T_ + t) * DQK_;
    if (tid < DH_) {
      Qp[qk + tid] = (bf16)(q[(size_t)t * (NH_ * DQK_) + h * DQK_ + tid] * SC);
      Kp[qk + tid] = (bf16)(kv[(size_t)t * (NH_ * 2 * DH_) + h * 2 * DH_ + tid]);
      Vp[((size_t)h * T_ + t) * DH_ + tid] =
          (bf16)(kv[(size_t)t * (NH_ * 2 * DH_) + h * 2 * DH_ + DH_ + tid]);
    } else {
      const int i = tid - DH_;
      const int ih = i & 31;
      const bool up = i >= 32;
      const float c = cs[t * 32 + ih], s = sn[t * 32 + ih];
      const float xr = q[(size_t)t * (NH_ * DQK_) + h * DQK_ + DH_ + ih];
      const float xi = q[(size_t)t * (NH_ * DQK_) + h * DQK_ + DH_ + 32 + ih];
      Qp[qk + tid] = (bf16)((up ? (xr * s + xi * c) : (xr * c - xi * s)) * SC);
      const float kxr = kr[(size_t)t * DR_ + ih];
      const float kxi = kr[(size_t)t * DR_ + 32 + ih];
      Kp[qk + tid] = (bf16)(up ? (kxr * s + kxi * c) : (kxr * c - kxi * s));
    }
  }
}

// ---------------------------------------------------------------------------
// Flash attention: grid (NH, T/64), 128 threads = 4 waves; causal.
// Q/K tiles stream via async-to-LDS; V transposed through LDS; bf16 output.
// ---------------------------------------------------------------------------
#define SQK 200
#define SV  72
__global__ __launch_bounds__(128) void k_attn(const bf16* __restrict__ Qp,
                                              const bf16* __restrict__ Kp,
                                              const bf16* __restrict__ Vp,
                                              bf16* __restrict__ attnb) {
  const int h = blockIdx.x;
  const int qb = blockIdx.y;
  const int q0 = qb * 64;
  __shared__ bf16 Qs[64 * SQK];
  __shared__ bf16 Ks[64 * SQK];
  __shared__ bf16 Vt[128 * SV];
  __shared__ bf16 Ps[64 * SV];
  const bf16* Qh = Qp + (size_t)h * T_ * DQK_;
  const bf16* Kh = Kp + (size_t)h * T_ * DQK_;
  const bf16* Vh = Vp + (size_t)h * T_ * DH_;
  const int tid = threadIdx.x;
  const int w = tid >> 5, lane = tid & 31;
  const int cn = lane & 15, hi = lane >> 4;

#pragma unroll
  for (int p = 0; p < 12; ++p) {
    const int idx = tid + p * 128;
    const int r = idx / 24, sir = idx % 24;
    async_b128(Qh + (size_t)(q0 + r) * DQK_ + sir * 8, &Qs[r * SQK + sir * 8]);
  }

  v8f o[8];
#pragma unroll
  for (int i = 0; i < 8; ++i) o[i] = v8f{0, 0, 0, 0, 0, 0, 0, 0};
  float mrow[8], lrow[8];
#pragma unroll
  for (int r = 0; r < 8; ++r) { mrow[r] = -1e30f; lrow[r] = 0.f; }

  for (int kb = 0; kb <= qb; ++kb) {
    const int k0 = kb * 64;
    __syncthreads();
#pragma unroll
    for (int p = 0; p < 12; ++p) {
      const int idx = tid + p * 128;
      const int r = idx / 24, sir = idx % 24;
      async_b128(Kh + (size_t)(k0 + r) * DQK_ + sir * 8, &Ks[r * SQK + sir * 8]);
    }
#pragma unroll
    for (int p = 0; p < 8; ++p) {
      const int idx = tid + p * 128;
      const int r = idx >> 4, c8 = (idx & 15) * 8;
      uint4 v = *reinterpret_cast<const uint4*>(Vh + (size_t)(k0 + r) * DH_ + c8);
      const bf16* pv = reinterpret_cast<const bf16*>(&v);
#pragma unroll
      for (int j = 0; j < 8; ++j) Vt[(c8 + j) * SV + r] = pv[j];
    }
    wait_async();
    __syncthreads();

    v16bf aq[6];
#pragma unroll
    for (int kk = 0; kk < 6; ++kk)
      aq[kk] = load_frag_a(&Qs[(w * 16) * SQK + kk * 32], SQK);

    v8f s[4];
#pragma unroll
    for (int nt = 0; nt < 4; ++nt) {
      s[nt] = v8f{0, 0, 0, 0, 0, 0, 0, 0};
#pragma unroll
      for (int kk = 0; kk < 6; ++kk) {
        v16bf bk = load_frag_b(&Ks[(nt * 16) * SQK + kk * 32], SQK);
        s[nt] = wmma_bf16(aq[kk], bk, s[nt]);
      }
    }

    if (kb == qb) {
#pragma unroll
      for (int nt = 0; nt < 4; ++nt)
#pragma unroll
        for (int r = 0; r < 8; ++r) {
          const int qi = w * 16 + r + hi * 8;
          const int ki = nt * 16 + cn;
          if (ki > qi) s[nt][r] = -1e30f;
        }
    }

#pragma unroll
    for (int r = 0; r < 8; ++r) {
      float mx = -1e30f;
#pragma unroll
      for (int nt = 0; nt < 4; ++nt) mx = fmaxf(mx, s[nt][r]);
#pragma unroll
      for (int off = 8; off > 0; off >>= 1) mx = fmaxf(mx, __shfl_xor(mx, off, 32));
      const float mn = fmaxf(mrow[r], mx);
      const float sc = __expf(mrow[r] - mn);
      mrow[r] = mn;
      float rs = 0.f;
#pragma unroll
      for (int nt = 0; nt < 4; ++nt) {
        const float p = __expf(s[nt][r] - mn);
        s[nt][r] = p;
        rs += p;
      }
#pragma unroll
      for (int off = 8; off > 0; off >>= 1) rs += __shfl_xor(rs, off, 32);
      lrow[r] = lrow[r] * sc + rs;
#pragma unroll
      for (int no = 0; no < 8; ++no) o[no][r] *= sc;
#pragma unroll
      for (int nt = 0; nt < 4; ++nt)
        Ps[(w * 16 + r + hi * 8) * SV + nt * 16 + cn] = (bf16)s[nt][r];
    }

    v16bf ap[2];
#pragma unroll
    for (int kk = 0; kk < 2; ++kk)
      ap[kk] = load_frag_a(&Ps[(w * 16) * SV + kk * 32], SV);
#pragma unroll
    for (int no = 0; no < 8; ++no)
#pragma unroll
      for (int kk = 0; kk < 2; ++kk) {
        v16bf bv = load_frag_b(&Vt[(no * 16) * SV + kk * 32], SV);
        o[no] = wmma_bf16(ap[kk], bv, o[no]);
      }
  }

#pragma unroll
  for (int r = 0; r < 8; ++r) {
    const int qi = q0 + w * 16 + r + hi * 8;
    const float inv = 1.f / lrow[r];
#pragma unroll
    for (int no = 0; no < 8; ++no)
      attnb[(size_t)qi * (NH_ * DH_) + h * DH_ + no * 16 + cn] =
          (bf16)(o[no][r] * inv);
  }
}

// ---------------------------------------------------------------------------
// Router: logits = h2 @ Wr, softmax, top-2, renormalize.
// ---------------------------------------------------------------------------
__global__ __launch_bounds__(128) void k_router(const float* __restrict__ h2,
                                                const float* __restrict__ Wr,
                                                int* __restrict__ ti,
                                                float* __restrict__ tw) {
  const int t = blockIdx.x;
  float acc[NE_];
#pragma unroll
  for (int e = 0; e < NE_; ++e) acc[e] = 0.f;
  for (int d = threadIdx.x; d < D_; d += 128) {
    const float hv = h2[(size_t)t * D_ + d];
#pragma unroll
    for (int e = 0; e < NE_; ++e) acc[e] += hv * Wr[(size_t)d * NE_ + e];
  }
  __shared__ float red[128 * NE_];
#pragma unroll
  for (int e = 0; e < NE_; ++e) red[threadIdx.x * NE_ + e] = acc[e];
  __syncthreads();
  for (int s = 64; s > 0; s >>= 1) {
    if (threadIdx.x < s)
#pragma unroll
      for (int e = 0; e < NE_; ++e)
        red[threadIdx.x * NE_ + e] += red[(threadIdx.x + s) * NE_ + e];
    __syncthreads();
  }
  if (threadIdx.x == 0) {
    float lg[NE_], mx = -1e30f;
#pragma unroll
    for (int e = 0; e < NE_; ++e) { lg[e] = red[e]; mx = fmaxf(mx, lg[e]); }
    float sum = 0.f;
#pragma unroll
    for (int e = 0; e < NE_; ++e) { lg[e] = __expf(lg[e] - mx); sum += lg[e]; }
#pragma unroll
    for (int e = 0; e < NE_; ++e) lg[e] /= sum;
    int i0 = 0;
#pragma unroll
    for (int e = 1; e < NE_; ++e) if (lg[e] > lg[i0]) i0 = e;
    int i1 = (i0 == 0) ? 1 : 0;
#pragma unroll
    for (int e = 0; e < NE_; ++e)
      if (e != i0 && lg[e] > lg[i1]) i1 = e;
    const float w0 = lg[i0], w1 = lg[i1], ws = w0 + w1;
    ti[t * 2 + 0] = i0; ti[t * 2 + 1] = i1;
    tw[t * 2 + 0] = w0 / ws; tw[t * 2 + 1] = w1 / ws;
  }
}

__global__ void k_moe_zero(int* c) { if (threadIdx.x < 3 * NE_) c[threadIdx.x] = 0; }

__global__ void k_moe_hist(const int* __restrict__ ti, int* __restrict__ c) {
  const int i = blockIdx.x * blockDim.x + threadIdx.x;
  if (i < NSLOT_) atomicAdd(&c[ti[i]], 1);
}

__global__ void k_moe_scan(int* c) {
  if (threadIdx.x == 0) {
    int off = 0;
    for (int e = 0; e < NE_; ++e) { c[NE_ + e] = off; off += c[e]; }
  }
}

__global__ void k_moe_build(const int* __restrict__ ti, const float* __restrict__ tw,
                            int* __restrict__ c, int* __restrict__ tok,
                            float* __restrict__ wl) {
  const int i = blockIdx.x * blockDim.x + threadIdx.x;
  if (i < NSLOT_) {
    const int e = ti[i];
    const int p = atomicAdd(&c[2 * NE_ + e], 1);
    const int s = c[NE_ + e] + p;
    tok[s] = i >> 1;
    wl[s] = tw[i];
  }
}

__global__ void k_silu_mul(const float* __restrict__ G, const float* __restrict__ U,
                           bf16* __restrict__ O, int n) {
  const int i = blockIdx.x * blockDim.x + threadIdx.x;
  if (i < n) {
    const float g = G[i];
    O[i] = (bf16)((g / (1.f + __expf(-g))) * U[i]);
  }
}

// ---------------------------------------------------------------------------
// Host orchestration
// ---------------------------------------------------------------------------
static inline dim3 gemm_grid(int M, int N) {
  return dim3((N + 127) / 128, (M + 127) / 128);
}

extern "C" void kernel_launch(void* const* d_in, const int* in_sizes, int n_in,
                              void* d_out, int out_size, void* d_ws, size_t ws_size,
                              hipStream_t stream) {
  const float* x     = (const float*)d_in[0];
  const float* W_dq  = (const float*)d_in[1];
  const float* W_uq  = (const float*)d_in[2];
  const float* W_dkv = (const float*)d_in[3];
  const float* W_ukv = (const float*)d_in[4];
  const float* W_kr  = (const float*)d_in[5];
  const float* W_o   = (const float*)d_in[6];
  const float* g_ln1 = (const float*)d_in[7];
  const float* g_ln2 = (const float*)d_in[8];
  const float* g_cq  = (const float*)d_in[9];
  const float* g_ckv = (const float*)d_in[10];
  const float* Wr    = (const float*)d_in[11];
  const float* Wg    = (const float*)d_in[12];
  const float* Wu    = (const float*)d_in[13];
  const float* Wd    = (const float*)d_in[14];
  const float* Wsg   = (const float*)d_in[15];
  const float* Wsu   = (const float*)d_in[16];
  const float* Wsd   = (const float*)d_in[17];
  const float* cosb  = (const float*)d_in[18];
  const float* sinb  = (const float*)d_in[19];
  float* out = (float*)d_out;

  char* ws = (char*)d_ws;
  size_t off = 0;
  auto alloc = [&](size_t bytes) -> char* {
    char* p = ws + off;
    off = (off + bytes + 255) & ~(size_t)255;
    return p;
  };
  // activations
  bf16*  hb    = (bf16*)alloc((size_t)T_ * D_ * 2);
  float* cqr   = (float*)alloc((size_t)T_ * CQ_ * 4);
  bf16*  cqb   = (bf16*)alloc((size_t)T_ * CQ_ * 2);
  float* q     = (float*)alloc((size_t)T_ * NH_ * DQK_ * 4);
  float* ckvr  = (float*)alloc((size_t)T_ * CKV_ * 4);
  bf16*  ckvb  = (bf16*)alloc((size_t)T_ * CKV_ * 2);
  float* kv    = (float*)alloc((size_t)T_ * NH_ * 2 * DH_ * 4);
  float* kr    = (float*)alloc((size_t)T_ * DR_ * 4);
  bf16*  Qp    = (bf16*)alloc((size_t)NH_ * T_ * DQK_ * 2);
  bf16*  Kp    = (bf16*)alloc((size_t)NH_ * T_ * DQK_ * 2);
  bf16*  Vp    = (bf16*)alloc((size_t)NH_ * T_ * DH_ * 2);
  bf16*  attnb = (bf16*)alloc((size_t)T_ * D_ * 2);
  float* x1    = (float*)alloc((size_t)T_ * D_ * 4);
  float* h2    = (float*)alloc((size_t)T_ * D_ * 4);
  bf16*  h2b   = (bf16*)alloc((size_t)T_ * D_ * 2);
  int*   ti    = (int*)alloc((size_t)NSLOT_ * 4);
  float* tw    = (float*)alloc((size_t)NSLOT_ * 4);
  int*   cnts  = (int*)alloc(3 * NE_ * 4);  // counts | offsets | cursors
  int*   tok   = (int*)alloc((size_t)NSLOT_ * 4);
  float* wl    = (float*)alloc((size_t)NSLOT_ * 4);
  float* G     = (float*)alloc((size_t)NSLOT_ * FE_ * 4);
  float* U     = (float*)alloc((size_t)NSLOT_ * FE_ * 4);
  bf16*  gactb = (bf16*)alloc((size_t)NSLOT_ * FE_ * 2);
  float* Gs    = (float*)alloc((size_t)T_ * FE_ * 4);
  float* Us    = (float*)alloc((size_t)T_ * FE_ * 4);
  bf16*  gsb   = (bf16*)alloc((size_t)T_ * FE_ * 2);
  // bf16 weight copies
  bf16* W_dqb  = (bf16*)alloc((size_t)D_ * CQ_ * 2);
  bf16* W_uqb  = (bf16*)alloc((size_t)CQ_ * NH_ * DQK_ * 2);
  bf16* W_dkvb = (bf16*)alloc((size_t)D_ * CKV_ * 2);
  bf16* W_ukvb = (bf16*)alloc((size_t)CKV_ * NH_ * 2 * DH_ * 2);
  bf16* W_krb  = (bf16*)alloc((size_t)D_ * DR_ * 2);
  bf16* W_ob   = (bf16*)alloc((size_t)NH_ * DH_ * D_ * 2);
  bf16* Wgb    = (bf16*)alloc((size_t)NE_ * D_ * FE_ * 2);
  bf16* Wub    = (bf16*)alloc((size_t)NE_ * D_ * FE_ * 2);
  bf16* Wdb    = (bf16*)alloc((size_t)NE_ * FE_ * D_ * 2);
  bf16* Wsgb   = (bf16*)alloc((size_t)D_ * FE_ * 2);
  bf16* Wsub   = (bf16*)alloc((size_t)D_ * FE_ * 2);
  bf16* Wsdb   = (bf16*)alloc((size_t)FE_ * D_ * 2);
  (void)ws_size; (void)in_sizes; (void)n_in; (void)out_size;

  auto conv = [&](const float* src, bf16* dst, size_t n) {
    const int n4 = (int)(n / 4);
    k_f2bf<<<(n4 + 255) / 256, 256, 0, stream>>>(src, dst, n4);
  };
  conv(W_dq,  W_dqb,  (size_t)D_ * CQ_);
  conv(W_uq,  W_uqb,  (size_t)CQ_ * NH_ * DQK_);
  conv(W_dkv, W_dkvb, (size_t)D_ * CKV_);
  conv(W_ukv, W_ukvb, (size_t)CKV_ * NH_ * 2 * DH_);
  conv(W_kr,  W_krb,  (size_t)D_ * DR_);
  conv(W_o,   W_ob,   (size_t)NH_ * DH_ * D_);
  conv(Wg,    Wgb,    (size_t)NE_ * D_ * FE_);
  conv(Wu,    Wub,    (size_t)NE_ * D_ * FE_);
  conv(Wd,    Wdb,    (size_t)NE_ * FE_ * D_);
  conv(Wsg,   Wsgb,   (size_t)D_ * FE_);
  conv(Wsu,   Wsub,   (size_t)D_ * FE_);
  conv(Wsd,   Wsdb,   (size_t)FE_ * D_);

#define GEMM(Ab, Bb, addp, Cp, Mv, Nv, Kv) \
  k_gemm_bf<<<gemm_grid(Mv, Nv), 256, 0, stream>>>(Ab, Bb, addp, Cp, Mv, Nv, Kv, \
                                                   nullptr, nullptr, nullptr,   \
                                                   nullptr, 0)

  // 1) h = rms(x, g_ln1) -> bf16
  k_rms<<<T_, 256, 0, stream>>>(x, g_ln1, nullptr, hb, D_);
  // 2) cq = rms(h @ W_dq, g_cq) -> bf16
  GEMM(hb, W_dqb, nullptr, cqr, T_, CQ_, D_);
  k_rms<<<T_, 256, 0, stream>>>(cqr, g_cq, nullptr, cqb, CQ_);
  // 3) q = cq @ W_uq
  GEMM(cqb, W_uqb, nullptr, q, T_, NH_ * DQK_, CQ_);
  // 4) ckv = rms(h @ W_dkv, g_ckv) -> bf16 ; kv = ckv @ W_ukv
  GEMM(hb, W_dkvb, nullptr, ckvr, T_, CKV_, D_);
  k_rms<<<T_, 256, 0, stream>>>(ckvr, g_ckv, nullptr, ckvb, CKV_);
  GEMM(ckvb, W_ukvb, nullptr, kv, T_, NH_ * 2 * DH_, CKV_);
  // 5) kr = h @ W_kr
  GEMM(hb, W_krb, nullptr, kr, T_, DR_, D_);
  // 6) pack bf16 Q/K/V per head (RoPE + scale)
  k_pack<<<T_, 256, 0, stream>>>(q, kv, kr, cosb, sinb, Qp, Kp, Vp);
  // 7) flash attention -> bf16
  k_attn<<<dim3(NH_, T_ / 64), 128, 0, stream>>>(Qp, Kp, Vp, attnb);
  // 8) x1 = x + attn @ W_o
  GEMM(attnb, W_ob, x, x1, T_, D_, D_);
  // 9) h2 = rms(x1, g_ln2) (fp32 for router + bf16 for GEMMs)
  k_rms<<<T_, 256, 0, stream>>>(x1, g_ln2, h2, h2b, D_);
  // 10) router + token scatter
  k_router<<<T_, 128, 0, stream>>>(h2, Wr, ti, tw);
  k_moe_zero<<<1, 64, 0, stream>>>(cnts);
  k_moe_hist<<<(NSLOT_ + 255) / 256, 256, 0, stream>>>(ti, cnts);
  k_moe_scan<<<1, 32, 0, stream>>>(cnts);
  k_moe_build<<<(NSLOT_ + 255) / 256, 256, 0, stream>>>(ti, tw, cnts, tok, wl);
  // 11) MoE up/gate (gather by token), activation -> bf16
  k_gemm_bf<<<dim3(FE_ / 128, 16, NE_), 256, 0, stream>>>(
      h2b, Wgb, nullptr, G, T_, FE_, D_, tok, wl, cnts, cnts + NE_, 1);
  k_gemm_bf<<<dim3(FE_ / 128, 16, NE_), 256, 0, stream>>>(
      h2b, Wub, nullptr, U, T_, FE_, D_, tok, wl, cnts, cnts + NE_, 1);
  k_silu_mul<<<(NSLOT_ * FE_ + 255) / 256, 256, 0, stream>>>(G, U, gactb,
                                                             NSLOT_ * FE_);
  // 12) shared expert
  GEMM(h2b, Wsgb, nullptr, Gs, T_, FE_, D_);
  GEMM(h2b, Wsub, nullptr, Us, T_, FE_, D_);
  k_silu_mul<<<(T_ * FE_ + 255) / 256, 256, 0, stream>>>(Gs, Us, gsb, T_ * FE_);
  // 13) out = x1 + shared   (fully overwrites d_out)
  GEMM(gsb, Wsdb, x1, out, T_, D_, FE_);
  // 14) out += MoE down-proj (scatter with routing weights)
  k_gemm_bf<<<dim3(D_ / 128, 16, NE_), 256, 0, stream>>>(
      gactb, Wdb, nullptr, out, T_, D_, FE_, tok, wl, cnts, cnts + NE_, 2);
#undef GEMM
}
